// BertCNN_50096498540917
// MI455X (gfx1250) — matrix operand order repaired and checked by reading, not compiled, not run on hardware
//
#include <hip/hip_runtime.h>
#include <hip/hip_bf16.h>
#include <math.h>

typedef __attribute__((ext_vector_type(16))) _Float16 v16h;
typedef __attribute__((ext_vector_type(8)))  _Float16 v8h;
typedef __attribute__((ext_vector_type(8)))  float    v8f;

union U16 { v16h v; v8h h[2]; };

// ---- model dims ----
constexpr int cD  = 512;
constexpr int cH  = 8;
constexpr int cDH = 64;
constexpr int cNL = 4;
constexpr int cFF = 2048;
constexpr int cNF = 256;
constexpr int cNC = 4;
constexpr int cB  = 32;
constexpr int cS  = 256;
constexpr int cM  = cB * cS;      // 8192 token rows
constexpr int cS1 = cS + 1;       // 257 padded rows/batch for conv
constexpr int cMc = cB * cS1;     // 8224 conv rows

// =====================================================================
// Weight prep: fp32 [R][C] -> f16 transposed [C][R]  (Bt layout for WMMA)
// =====================================================================
__global__ void transpose_to_f16(const float* __restrict__ W,
                                 _Float16* __restrict__ Wt, int R, int C) {
  int idx = blockIdx.x * 256 + threadIdx.x;
  if (idx < R * C) {
    int r = idx / C, c = idx % C;
    Wt[(size_t)c * R + r] = (_Float16)W[idx];
  }
}

__global__ void convert_f16(const float* __restrict__ W,
                            _Float16* __restrict__ O, int n) {
  int idx = blockIdx.x * 256 + threadIdx.x;
  if (idx < n) O[idx] = (_Float16)W[idx];
}

// =====================================================================
// Generic WMMA GEMM:  C[M][N] = act( A[M][K](f16) * Bt[N][K](f16)^T
//                                    + bias[N] + res[M][N] )
// BM=128 BN=128 BK=32, 256 threads = 8 waves, each wave -> 32x64 outputs.
// Staging: double-buffered LDS filled by global_load_async_to_lds_b128
// (ASYNCcnt DMA, no VGPR round-trip); tile i+1 DMA overlaps tile i WMMAs.
// Out-of-range rows are clamped (not branched): their products only land
// in output elements that the epilogue discards.
// =====================================================================
__global__ __launch_bounds__(256) void gemm_wmma(
    const _Float16* __restrict__ A, int lda,
    const _Float16* __restrict__ Bt, int ldb,
    const float* __restrict__ bias,
    const float* __restrict__ res,
    float* __restrict__ Cf, _Float16* __restrict__ Ch, int ldc,
    int M, int N, int K, int act) {
  __shared__ _Float16 sA[2][128][32];   // 16 KB double-buffered
  __shared__ _Float16 sB[2][128][32];   // 16 KB (Bt tile: [n][k])

  const int tid  = threadIdx.x;
  const int bm   = blockIdx.y * 128;
  const int bn   = blockIdx.x * 128;
  const int wave = tid >> 5;
  const int lane = tid & 31;
  const int m16  = lane & 15;
  const int hig  = lane >> 4;
  const int wr   = (wave & 3) * 32;   // wave row offset in tile
  const int wc   = (wave >> 2) * 64;  // wave col offset in tile

  v8f acc[2][4];
  for (int i = 0; i < 2; i++)
    for (int j = 0; j < 4; j++)
      for (int e = 0; e < 8; e++) acc[i][j][e] = 0.f;

  const int ar = (tid * 16) >> 5;   // 0..127 (tile row this thread stages)
  const int ac = (tid * 16) & 31;   // 0 or 16

  // clamped global source rows (branch-free staging)
  int grA = bm + ar; if (grA > M - 1) grA = M - 1;
  int gnB = bn + ar; if (gnB > N - 1) gnB = N - 1;
  const _Float16* gA = &A[(size_t)grA * lda + ac];
  const _Float16* gB = &Bt[(size_t)gnB * ldb + ac];
  // LDS byte offsets (flat->LDS mapping uses addr[31:0])
  unsigned ldsA[2] = { (unsigned)(size_t)&sA[0][ar][ac],
                       (unsigned)(size_t)&sA[1][ar][ac] };
  unsigned ldsB[2] = { (unsigned)(size_t)&sB[0][ar][ac],
                       (unsigned)(size_t)&sB[1][ar][ac] };

  auto stage = [&](int buf, int k0) {
    unsigned long long aAddr = (unsigned long long)(size_t)(gA + k0);
    unsigned long long bAddr = (unsigned long long)(size_t)(gB + k0);
    asm volatile("global_load_async_to_lds_b128 %0, %1, off\n\t"
                 "global_load_async_to_lds_b128 %0, %1, off offset:16"
                 :: "v"(ldsA[buf]), "v"(aAddr) : "memory");
    asm volatile("global_load_async_to_lds_b128 %0, %1, off\n\t"
                 "global_load_async_to_lds_b128 %0, %1, off offset:16"
                 :: "v"(ldsB[buf]), "v"(bAddr) : "memory");
  };

  const int nk = K >> 5;
  stage(0, 0);
  asm volatile("s_wait_asynccnt 0" ::: "memory");
  __syncthreads();

  for (int ki = 0; ki < nk; ki++) {
    const int buf = ki & 1;
    if (ki + 1 < nk) stage(buf ^ 1, (ki + 1) << 5);   // DMA next tile

    // ISA 7.12.2 16-bit A fragment: lanes 0-15 K{0-7,16-23}, lanes 16-31 K{8-15,24-31}
    U16 fa[2];
    for (int i = 0; i < 2; i++) {
      const _Float16* p = &sA[buf][wr + i * 16 + m16][hig * 8];
      fa[i].h[0] = *(const v8h*)p;
      fa[i].h[1] = *(const v8h*)(p + 16);
    }
    // B fragment from Bt tile: lane col = m16, same K order as A fragment
    U16 fb[4];
    for (int j = 0; j < 4; j++) {
      const _Float16* p = &sB[buf][wc + j * 16 + m16][hig * 8];
      fb[j].h[0] = *(const v8h*)p;
      fb[j].h[1] = *(const v8h*)(p + 16);
    }
    for (int i = 0; i < 2; i++)
      for (int j = 0; j < 4; j++)
        acc[i][j] = __builtin_amdgcn_wmma_f32_16x16x32_f16(
            false, fa[i].v, false, fb[j].v, (short)0, acc[i][j], false, false);

    asm volatile("s_wait_asynccnt 0" ::: "memory");  // next tile landed
    __syncthreads();                                  // all waves done w/ buf
  }

  // epilogue: bias + residual + activation, write f32 and/or f16
  for (int i = 0; i < 2; i++) {
    int rbase = bm + wr + i * 16 + hig * 8;
    for (int j = 0; j < 4; j++) {
      int n = bn + wc + j * 16 + m16;
      if (n >= N) continue;
      float bia = bias ? bias[n] : 0.f;
      for (int e = 0; e < 8; e++) {
        int m = rbase + e;
        if (m >= M) continue;
        float val = acc[i][j][e] + bia;
        if (res) val += res[(size_t)m * ldc + n];
        if (act == 1) val = 0.5f * val * (1.f + erff(val * 0.70710678118f)); // exact GELU
        else if (act == 2) val = fmaxf(val, 0.f);                            // ReLU
        if (Cf) Cf[(size_t)m * ldc + n] = val;
        if (Ch) Ch[(size_t)m * ldc + n] = (_Float16)val;
      }
    }
  }
}

// =====================================================================
// Embedding + LayerNorm (one block per token row, D=512, 2 elems/thread)
// =====================================================================
__global__ __launch_bounds__(256) void embed_ln(
    const int* __restrict__ ids, const int* __restrict__ tids,
    const float* __restrict__ we, const float* __restrict__ pe,
    const float* __restrict__ te, const float* __restrict__ g,
    const float* __restrict__ bta, float* __restrict__ xo,
    _Float16* __restrict__ xh) {
  __shared__ float red[256];
  int row = blockIdx.x;          // b*S + s
  int s   = row % cS;
  int t   = threadIdx.x;
  int id  = ids[row], tt = tids[row];
  float v0 = we[(size_t)id * cD + t]       + pe[s * cD + t]       + te[tt * cD + t];
  float v1 = we[(size_t)id * cD + t + 256] + pe[s * cD + t + 256] + te[tt * cD + t + 256];

  red[t] = v0 + v1; __syncthreads();
  for (int o = 128; o > 0; o >>= 1) { if (t < o) red[t] += red[t + o]; __syncthreads(); }
  float mean = red[0] * (1.f / 512.f); __syncthreads();
  float d0 = v0 - mean, d1 = v1 - mean;
  red[t] = d0 * d0 + d1 * d1; __syncthreads();
  for (int o = 128; o > 0; o >>= 1) { if (t < o) red[t] += red[t + o]; __syncthreads(); }
  float inv = rsqrtf(red[0] * (1.f / 512.f) + 1e-12f);
  float y0 = d0 * inv * g[t] + bta[t];
  float y1 = d1 * inv * g[t + 256] + bta[t + 256];
  xo[(size_t)row * cD + t] = y0;        xo[(size_t)row * cD + t + 256] = y1;
  xh[(size_t)row * cD + t] = (_Float16)y0; xh[(size_t)row * cD + t + 256] = (_Float16)y1;
}

// LayerNorm of a precomputed f32 buffer -> f32 + f16
__global__ __launch_bounds__(256) void ln_kernel(
    const float* __restrict__ x, const float* __restrict__ g,
    const float* __restrict__ bta, float* __restrict__ xo,
    _Float16* __restrict__ xh) {
  __shared__ float red[256];
  int row = blockIdx.x;
  int t = threadIdx.x;
  float v0 = x[(size_t)row * cD + t];
  float v1 = x[(size_t)row * cD + t + 256];
  red[t] = v0 + v1; __syncthreads();
  for (int o = 128; o > 0; o >>= 1) { if (t < o) red[t] += red[t + o]; __syncthreads(); }
  float mean = red[0] * (1.f / 512.f); __syncthreads();
  float d0 = v0 - mean, d1 = v1 - mean;
  red[t] = d0 * d0 + d1 * d1; __syncthreads();
  for (int o = 128; o > 0; o >>= 1) { if (t < o) red[t] += red[t + o]; __syncthreads(); }
  float inv = rsqrtf(red[0] * (1.f / 512.f) + 1e-12f);
  float y0 = d0 * inv * g[t] + bta[t];
  float y1 = d1 * inv * g[t + 256] + bta[t + 256];
  xo[(size_t)row * cD + t] = y0;        xo[(size_t)row * cD + t + 256] = y1;
  xh[(size_t)row * cD + t] = (_Float16)y0; xh[(size_t)row * cD + t + 256] = (_Float16)y1;
}

// =====================================================================
// Attention: one block per (b, h, half). 8 waves, 16 q-rows per wave.
// Q.K^T via WMMA from global f16, softmax in LDS, P.V via WMMA from LDS.
// Dynamic LDS: sS 128x256 f32 | sVt 64x256 f16 | sP 128x256 f16 = 224 KB
// =====================================================================
__global__ __launch_bounds__(256) void attn_kernel(
    const _Float16* __restrict__ q, const _Float16* __restrict__ k,
    const _Float16* __restrict__ v, const int* __restrict__ mask,
    _Float16* __restrict__ ctx) {
  extern __shared__ char smem[];
  float*    sS  = (float*)smem;                             // 131072 B
  _Float16* sVt = (_Float16*)(smem + 131072);               //  32768 B
  _Float16* sP  = (_Float16*)(smem + 131072 + 32768);       //  65536 B

  const int half = blockIdx.x, h = blockIdx.y, b = blockIdx.z;
  const int tid = threadIdx.x, wave = tid >> 5, lane = tid & 31;
  const int m16 = lane & 15, hig = lane >> 4;

  // transpose V(b,h) 256x64 -> sVt[64][256]
  {
    const _Float16* src = &v[((size_t)(b * cS + tid)) * cD + h * cDH];
    for (int d0 = 0; d0 < cDH; d0 += 8) {
      v8h c8 = *(const v8h*)(src + d0);
      for (int e = 0; e < 8; e++) sVt[(d0 + e) * cS + tid] = c8[e];
    }
  }
  __syncthreads();

  const int qr = half * 128 + wave * 16;   // first q row of this wave
  // scores = Q.K^T / 8 + mask bias
  for (int kt = 0; kt < 16; kt++) {
    v8f acc; for (int e = 0; e < 8; e++) acc[e] = 0.f;
    for (int ks = 0; ks < cDH; ks += 32) {
      U16 fa, fb;
      const _Float16* pa = &q[((size_t)(b * cS + qr + m16)) * cD + h * cDH + ks + hig * 8];
      fa.h[0] = *(const v8h*)pa;
      fa.h[1] = *(const v8h*)(pa + 16);
      const _Float16* pb = &k[((size_t)(b * cS + kt * 16 + m16)) * cD + h * cDH + ks + hig * 8];
      fb.h[0] = *(const v8h*)pb;
      fb.h[1] = *(const v8h*)(pb + 16);
      acc = __builtin_amdgcn_wmma_f32_16x16x32_f16(false, fa.v, false, fb.v,
                                                   (short)0, acc, false, false);
    }
    int col = kt * 16 + m16;
    float ab = ((float)mask[b * cS + col] - 1.f) * 10000.f;
    int lr = wave * 16 + hig * 8;
    for (int e = 0; e < 8; e++)
      sS[(lr + e) * cS + col] = acc[e] * 0.125f + ab;
  }
  __syncthreads();

  // softmax: one thread per local row
  if (tid < 128) {
    float* row = &sS[tid * cS];
    float mx = -1e30f;
    for (int c = 0; c < cS; c++) mx = fmaxf(mx, row[c]);
    float sum = 0.f;
    for (int c = 0; c < cS; c++) { float e = __expf(row[c] - mx); row[c] = e; sum += e; }
    float inv = 1.f / sum;
    _Float16* prow = &sP[tid * cS];
    for (int c = 0; c < cS; c++) prow[c] = (_Float16)(row[c] * inv);
  }
  __syncthreads();

  // ctx = P.V  (K-dim = 256)
  for (int dt = 0; dt < 4; dt++) {
    v8f acc; for (int e = 0; e < 8; e++) acc[e] = 0.f;
    for (int k0 = 0; k0 < cS; k0 += 32) {
      U16 fa, fb;
      const _Float16* pa = &sP[(wave * 16 + m16) * cS + k0 + hig * 8];
      fa.h[0] = *(const v8h*)pa;
      fa.h[1] = *(const v8h*)(pa + 16);
      const _Float16* pb = &sVt[(dt * 16 + m16) * cS + k0 + hig * 8];
      fb.h[0] = *(const v8h*)pb;
      fb.h[1] = *(const v8h*)(pb + 16);
      acc = __builtin_amdgcn_wmma_f32_16x16x32_f16(false, fa.v, false, fb.v,
                                                   (short)0, acc, false, false);
    }
    for (int e = 0; e < 8; e++) {
      int row = b * cS + half * 128 + wave * 16 + e + hig * 8;
      int col = h * cDH + dt * 16 + m16;
      ctx[(size_t)row * cD + col] = (_Float16)acc[e];
    }
  }
}

// =====================================================================
// Conv prep: masked, zero-padded f16 embedding [B][257][D] (+4 zero rows)
// =====================================================================
__global__ __launch_bounds__(256) void embp_kernel(
    const float* __restrict__ x, const int* __restrict__ mask,
    _Float16* __restrict__ embp) {
  int r = blockIdx.x;            // 0 .. cMc+3
  int t = threadIdx.x;
  int b = r / cS1, s = r % cS1;
  bool valid = (r < cMc) && (s < cS);
  float m = valid ? (float)mask[b * cS + s] : 0.f;
  float v0 = valid ? x[((size_t)(b * cS + s)) * cD + t] * m : 0.f;
  float v1 = valid ? x[((size_t)(b * cS + s)) * cD + t + 256] * m : 0.f;
  embp[(size_t)r * cD + t]       = (_Float16)v0;
  embp[(size_t)r * cD + t + 256] = (_Float16)v1;
}

// masked max-pool over positions for one conv set
__global__ __launch_bounds__(256) void pool_kernel(
    const float* __restrict__ co, const int* __restrict__ mask,
    float* __restrict__ rep, int kk, int repOff) {
  int idx = blockIdx.x * 256 + threadIdx.x;   // b*256 + f
  if (idx >= cB * cNF) return;
  int b = idx >> 8, f = idx & 255;
  int len = 0;
  for (int s = 0; s < cS; s++) len += mask[b * cS + s];
  int nw = cS1 - kk + 1;
  float best = -1e30f;
  for (int s = 0; s < nw; s++)
    if (s + kk - 1 <= len)
      best = fmaxf(best, co[((size_t)(b * cS1 + s)) * cNF + f]);
  rep[b * (3 * cNF * 3) + repOff + (kk - 1) * cNF + f] = best;
}

// final FC: out[b][c] = rep[b][:] . fc_w[:][c] + fc_b[c]
__global__ __launch_bounds__(128) void fc_kernel(
    const float* __restrict__ rep, const float* __restrict__ fw,
    const float* __restrict__ fb, float* __restrict__ out) {
  int t = threadIdx.x;
  if (t >= cB * cNC) return;
  int b = t >> 2, c = t & 3;
  float acc = fb[c];
  for (int i = 0; i < 3 * cNF * 3; i++)
    acc += rep[b * (3 * cNF * 3) + i] * fw[i * cNC + c];
  out[b * cNC + c] = acc;
}

// =====================================================================
extern "C" void kernel_launch(void* const* d_in, const int* in_sizes, int n_in,
                              void* d_out, int out_size, void* d_ws, size_t ws_size,
                              hipStream_t stream) {
  (void)in_sizes; (void)n_in; (void)out_size; (void)ws_size;
  // ---- inputs ----
  const int* q_ids  = (const int*)d_in[0];
  const int* q_tt   = (const int*)d_in[1];
  const int* q_msk  = (const int*)d_in[2];
  const int* a_ids  = (const int*)d_in[3];
  const int* a_tt   = (const int*)d_in[4];
  const int* a_msk  = (const int*)d_in[5];
  const int* b_ids  = (const int*)d_in[6];
  const int* b_tt   = (const int*)d_in[7];
  const int* b_msk  = (const int*)d_in[8];
  const float* word_emb = (const float*)d_in[9];
  const float* pos_emb  = (const float*)d_in[10];
  const float* type_emb = (const float*)d_in[11];
  const float* eln_s = (const float*)d_in[12];
  const float* eln_b = (const float*)d_in[13];
  const float* Wq = (const float*)d_in[14];  const float* bq = (const float*)d_in[15];
  const float* Wk = (const float*)d_in[16];  const float* bk = (const float*)d_in[17];
  const float* Wv = (const float*)d_in[18];  const float* bv = (const float*)d_in[19];
  const float* Wo = (const float*)d_in[20];  const float* bo = (const float*)d_in[21];
  const float* ln1s = (const float*)d_in[22]; const float* ln1b = (const float*)d_in[23];
  const float* Wi = (const float*)d_in[24];  const float* bi = (const float*)d_in[25];
  const float* Wo2 = (const float*)d_in[26]; const float* bo2 = (const float*)d_in[27];
  const float* ln2s = (const float*)d_in[28]; const float* ln2b = (const float*)d_in[29];
  const float* cw1 = (const float*)d_in[30]; const float* cb1 = (const float*)d_in[31];
  const float* cw2 = (const float*)d_in[32]; const float* cb2 = (const float*)d_in[33];
  const float* cw3 = (const float*)d_in[34]; const float* cb3 = (const float*)d_in[35];
  const float* fcw = (const float*)d_in[36]; const float* fcb = (const float*)d_in[37];
  float* out = (float*)d_out;

  // ---- workspace carve ----
  char* w = (char*)d_ws;
  auto alloc = [&](size_t bytes) {
    char* p = w; w += (bytes + 255) & ~(size_t)255; return p;
  };
  _Float16* WtQ  = (_Float16*)alloc((size_t)cNL * cD * cD * 2);
  _Float16* WtK  = (_Float16*)alloc((size_t)cNL * cD * cD * 2);
  _Float16* WtV  = (_Float16*)alloc((size_t)cNL * cD * cD * 2);
  _Float16* WtO  = (_Float16*)alloc((size_t)cNL * cD * cD * 2);
  _Float16* WtI  = (_Float16*)alloc((size_t)cNL * cFF * cD * 2);
  _Float16* WtO2 = (_Float16*)alloc((size_t)cNL * cD * cFF * 2);
  _Float16* CW1  = (_Float16*)alloc((size_t)cNF * 1 * cD * 2);
  _Float16* CW2  = (_Float16*)alloc((size_t)cNF * 2 * cD * 2);
  _Float16* CW3  = (_Float16*)alloc((size_t)cNF * 3 * cD * 2);
  float*    x32  = (float*)   alloc((size_t)cM * cD * 4);
  _Float16* x16  = (_Float16*)alloc((size_t)cM * cD * 2);
  float*    tmp32= (float*)   alloc((size_t)cM * cD * 4);
  _Float16* qbuf = (_Float16*)alloc((size_t)cM * cD * 2);
  _Float16* kbuf = (_Float16*)alloc((size_t)cM * cD * 2);
  _Float16* vbuf = (_Float16*)alloc((size_t)cM * cD * 2);
  _Float16* ctx  = (_Float16*)alloc((size_t)cM * cD * 2);
  _Float16* hbuf = (_Float16*)alloc((size_t)cM * cFF * 2);
  _Float16* embp = (_Float16*)alloc((size_t)(cMc + 4) * cD * 2);
  float*    cout = (float*)   alloc((size_t)cMc * cNF * 4);
  float*    rep  = (float*)   alloc((size_t)cB * 3 * cNF * 3 * 4);

  // ---- weight prep (once per launch, deterministic) ----
  for (int l = 0; l < cNL; l++) {
    size_t dd = (size_t)l * cD * cD;
    size_t df = (size_t)l * cD * cFF;
    transpose_to_f16<<<(cD * cD + 255) / 256, 256, 0, stream>>>(Wq + dd, WtQ + dd, cD, cD);
    transpose_to_f16<<<(cD * cD + 255) / 256, 256, 0, stream>>>(Wk + dd, WtK + dd, cD, cD);
    transpose_to_f16<<<(cD * cD + 255) / 256, 256, 0, stream>>>(Wv + dd, WtV + dd, cD, cD);
    transpose_to_f16<<<(cD * cD + 255) / 256, 256, 0, stream>>>(Wo + dd, WtO + dd, cD, cD);
    transpose_to_f16<<<(cD * cFF + 255) / 256, 256, 0, stream>>>(Wi + df, WtI + df, cD, cFF);
    transpose_to_f16<<<(cD * cFF + 255) / 256, 256, 0, stream>>>(Wo2 + df, WtO2 + df, cFF, cD);
  }
  convert_f16<<<(cNF * 1 * cD + 255) / 256, 256, 0, stream>>>(cw1, CW1, cNF * 1 * cD);
  convert_f16<<<(cNF * 2 * cD + 255) / 256, 256, 0, stream>>>(cw2, CW2, cNF * 2 * cD);
  convert_f16<<<(cNF * 3 * cD + 255) / 256, 256, 0, stream>>>(cw3, CW3, cNF * 3 * cD);

  auto gemm = [&](const _Float16* A, int lda, const _Float16* Bt, int ldb,
                  const float* bias, const float* res, float* Cf, _Float16* Ch,
                  int ldc, int M, int N, int K, int act) {
    dim3 g((N + 127) / 128, (M + 127) / 128);
    gemm_wmma<<<g, 256, 0, stream>>>(A, lda, Bt, ldb, bias, res, Cf, Ch, ldc, M, N, K, act);
  };

  const _Float16* convW[3] = {CW1, CW2, CW3};
  const float*    convB[3] = {cb1, cb2, cb3};

  auto encode = [&](const int* ids, const int* tids, const int* mask, int repOff) {
    embed_ln<<<cM, 256, 0, stream>>>(ids, tids, word_emb, pos_emb, type_emb,
                                     eln_s, eln_b, x32, x16);
    for (int l = 0; l < cNL; l++) {
      size_t dd = (size_t)l * cD * cD;
      size_t df = (size_t)l * cD * cFF;
      // QKV projections -> f16
      gemm(x16, cD, WtQ + dd, cD, bq + l * cD, nullptr, nullptr, qbuf, cD, cM, cD, cD, 0);
      gemm(x16, cD, WtK + dd, cD, bk + l * cD, nullptr, nullptr, kbuf, cD, cM, cD, cD, 0);
      gemm(x16, cD, WtV + dd, cD, bv + l * cD, nullptr, nullptr, vbuf, cD, cM, cD, cD, 0);
      // attention
      dim3 ag(2, cH, cB);
      size_t shmem = 131072 + 32768 + 65536;   // 224 KB of the 320 KB WGP LDS
      attn_kernel<<<ag, 256, shmem, stream>>>(qbuf, kbuf, vbuf, mask, ctx);
      // output proj + residual, then LN1
      gemm(ctx, cD, WtO + dd, cD, bo + l * cD, x32, tmp32, nullptr, cD, cM, cD, cD, 0);
      ln_kernel<<<cM, 256, 0, stream>>>(tmp32, ln1s + l * cD, ln1b + l * cD, x32, x16);
      // FFN: GELU(x @ Wi + bi) -> f16 ; then @ Wo2 + residual, LN2
      gemm(x16, cD, WtI + df, cD, bi + l * cFF, nullptr, nullptr, hbuf, cFF, cM, cFF, cD, 1);
      gemm(hbuf, cFF, WtO2 + df, cFF, bo2 + l * cD, x32, tmp32, nullptr, cD, cM, cD, cFF, 0);
      ln_kernel<<<cM, 256, 0, stream>>>(tmp32, ln2s + l * cD, ln2b + l * cD, x32, x16);
    }
    // conv head: masked padded f16 emb, conv-as-GEMM per tap, relu on last tap
    embp_kernel<<<cMc + 4, 256, 0, stream>>>(x32, mask, embp);
    for (int kset = 0; kset < 3; kset++) {
      int kk = kset + 1;
      for (int j = 0; j < kk; j++) {
        gemm(embp + (size_t)j * cD, cD,             // A shifted by tap
             convW[kset] + (size_t)j * cD, kk * cD, // Bt row n -> conv_w[n][j][:]
             (j == 0) ? convB[kset] : nullptr,
             (j == 0) ? nullptr : cout,             // accumulate via residual path
             cout, nullptr, cNF, cMc, cNF, cD,
             (j == kk - 1) ? 2 : 0);                // ReLU on final tap
      }
      pool_kernel<<<cB, 256, 0, stream>>>(cout, mask, rep, kk, repOff);
    }
  };

  // final concat order is (q_rep, b_rep, a_rep)
  encode(q_ids, q_tt, q_msk, 0);
  encode(b_ids, b_tt, b_msk, 768);
  encode(a_ids, a_tt, a_msk, 1536);

  fc_kernel<<<1, 128, 0, stream>>>(rep, fcw, fcb, out);
}
